// RecurrentRGCN_13494787244213
// MI455X (gfx1250) — matrix operand order repaired and checked by reading, not compiled, not run on hardware
//
#include <hip/hip_runtime.h>
#include <hip/hip_bf16.h>
#include <cstdint>
#include <math.h>

typedef __attribute__((ext_vector_type(16))) _Float16 v16h;
typedef __attribute__((ext_vector_type(8)))  _Float16 v8h;
typedef __attribute__((ext_vector_type(8)))  float    v8f;

#define HDIM 256
#define RRELU_SLOPE 0.2291666666666667f

__device__ __forceinline__ v8f wmma16(v16h a, v16h b, v8f c) {
  return __builtin_amdgcn_wmma_f32_16x16x32_f16(false, a, false, b, (short)0, c,
                                                false, false);
}

__device__ __forceinline__ float sigm(float x) { return 1.f / (1.f + __expf(-x)); }

// ---------------------------------------------------------------- utilities
__global__ void zero_kernel(float* p, long n) {
  long i = (long)blockIdx.x * blockDim.x + threadIdx.x;
  if (i < n) p[i] = 0.f;
}

__global__ void copy_kernel(float* dst, const float* src, long n) {
  long i = (long)blockIdx.x * blockDim.x + threadIdx.x;
  if (i < n) dst[i] = src[i];
}

// one block per row (256 threads); dst = l2norm(src) rows of 256
__global__ void l2norm_rows_kernel(float* dst, const float* src) {
  __shared__ float red[256];
  int r = blockIdx.x, c = threadIdx.x;
  float v = src[(size_t)r * HDIM + c];
  red[c] = v * v;
  __syncthreads();
  for (int s = 128; s > 0; s >>= 1) {
    if (c < s) red[c] += red[c + s];
    __syncthreads();
  }
  float invn = 1.f / fmaxf(sqrtf(red[0]), 1e-12f);
  dst[(size_t)r * HDIM + c] = v * invn;
}

// Pack a row-major [256][256] f32 matrix (optionally minus Wsub) into the
// wmma B-fragment layout: out[((nt*8+kt)*32 + lane)*16 + j] =
//   W[kt*32 + ((lane>>4)<<3) + (j&7) + ((j>>3)<<4)][nt*16 + (lane&15)]
__global__ void pack_b_kernel(const float* W, const float* Wsub, _Float16* out) {
  int idx = blockIdx.x * blockDim.x + threadIdx.x;
  if (idx >= HDIM * HDIM) return;
  int j = idx & 15;
  int lane = (idx >> 4) & 31;
  int kt = (idx >> 9) & 7;
  int nt = idx >> 12;
  int k = kt * 32 + ((lane >> 4) << 3) + (j & 7) + ((j >> 3) << 4);
  int n = nt * 16 + (lane & 15);
  float v = W[(size_t)k * HDIM + n];
  if (Wsub) v -= Wsub[(size_t)k * HDIM + n];
  out[idx] = (_Float16)v;
}

// ---------------------------------------------------------------- per-step ops
__global__ void rel_scatter_kernel(const int* rei, const int* rer, const float* h,
                                   float* xsum, float* cnt, int M) {
  long gid = (long)blockIdx.x * blockDim.x + threadIdx.x;
  if (gid >= (long)M * 64) return;
  int m = (int)(gid >> 6), q = (int)(gid & 63);
  int e = rei[m], r = rer[m];
  float4 v = *(const float4*)(h + (size_t)e * HDIM + q * 4);
  float* d = xsum + (size_t)r * HDIM + q * 4;
  atomicAdd(d + 0, v.x);
  atomicAdd(d + 1, v.y);
  atomicAdd(d + 2, v.z);
  atomicAdd(d + 3, v.w);
  if (q == 0) atomicAdd(cnt + r, 1.0f);
}

__global__ void indeg_kernel(const int* dst, float* indeg, int E) {
  int e = blockIdx.x * blockDim.x + threadIdx.x;
  if (e < E) atomicAdd(indeg + dst[e], 1.0f);
}

// GRU + l2norm over relation rows; h0 updated in place (row-local).
__global__ void gru_kernel(const float* emb_rel, const float* xsum, const float* cnt,
                           const float* w_ih, const float* w_hh,
                           const float* b_ih, const float* b_hh, float* h0) {
  __shared__ float red[256];
  int r = blockIdx.x, c = threadIdx.x;
  const float* er = emb_rel + (size_t)r * HDIM;
  const float* xs = xsum + (size_t)r * HDIM;
  float* h0r = h0 + (size_t)r * HDIM;
  float cv = cnt[r];
  float invc = cv > 0.f ? 1.f / cv : 0.f;
  const float* wr = w_ih + (size_t)c * (2 * HDIM);
  const float* wz = w_ih + (size_t)(c + HDIM) * (2 * HDIM);
  const float* wn = w_ih + (size_t)(c + 2 * HDIM) * (2 * HDIM);
  float ir = b_ih[c], iz = b_ih[c + HDIM], in_ = b_ih[c + 2 * HDIM];
  for (int k = 0; k < HDIM; k++) {
    float x = er[k];
    ir = fmaf(x, wr[k], ir);
    iz = fmaf(x, wz[k], iz);
    in_ = fmaf(x, wn[k], in_);
  }
  for (int k = 0; k < HDIM; k++) {
    float x = xs[k] * invc;
    ir = fmaf(x, wr[HDIM + k], ir);
    iz = fmaf(x, wz[HDIM + k], iz);
    in_ = fmaf(x, wn[HDIM + k], in_);
  }
  const float* vr = w_hh + (size_t)c * HDIM;
  const float* vz = w_hh + (size_t)(c + HDIM) * HDIM;
  const float* vn = w_hh + (size_t)(c + 2 * HDIM) * HDIM;
  float hr = b_hh[c], hz = b_hh[c + HDIM], hn = b_hh[c + 2 * HDIM];
  for (int k = 0; k < HDIM; k++) {
    float x = h0r[k];
    hr = fmaf(x, vr[k], hr);
    hz = fmaf(x, vz[k], hz);
    hn = fmaf(x, vn[k], hn);
  }
  float rg = sigm(ir + hr);
  float z = sigm(iz + hz);
  float n = tanhf(in_ + rg * hn);
  float out = (1.f - z) * n + z * h0r[c];
  red[c] = out * out;
  __syncthreads();
  for (int s = 128; s > 0; s >>= 1) {
    if (c < s) red[c] += red[c + s];
    __syncthreads();
  }
  float invn = 1.f / fmaxf(sqrtf(red[0]), 1e-12f);
  h0r[c] = out * invn;  // all reads of h0r happened before the barriers above
}

// ------------------------------------------------------------------ msg GEMM
// Block: 16 edges x 256 cols, 256 threads (8 waves x 2 n-tiles).
// A = f16(h[src]+h0[etype]) staged in LDS in fragment-permuted layout,
// D scattered with global f32 atomics into agg[dst].
__global__ void msg_kernel(const int* src, const int* dst, const int* etype,
                           const float* h, const float* h0,
                           const _Float16* __restrict__ wn_pack, float* agg, int E) {
  __shared__ __align__(32) _Float16 ldsA[16 * HDIM];
  __shared__ int s_s[16], s_e[16], s_d[16];
  int tid = threadIdx.x;
  int e0 = blockIdx.x * 16;
  if (tid < 16) {
    int e = e0 + tid;
    bool ok = e < E;
    s_s[tid] = ok ? src[e] : 0;
    s_e[tid] = ok ? etype[e] : 0;
    s_d[tid] = ok ? dst[e] : 0;
  }
  __syncthreads();
  for (int i = 0; i < 2; i++) {
    int task = tid * 2 + i;
    int row = task >> 5, g = task & 31;
    int kt = g >> 2, sub = g & 3;
    int kbase = kt * 32 + sub * 8;
    int hiflag = sub & 1, eoff = (sub >> 1) * 8;
    v8h o = {};
    if (e0 + row < E) {
      const float* ph = h + (size_t)s_s[row] * HDIM + kbase;
      const float* pr = h0 + (size_t)s_e[row] * HDIM + kbase;
      float4 a0 = *(const float4*)ph, a1 = *(const float4*)(ph + 4);
      float4 b0 = *(const float4*)pr, b1 = *(const float4*)(pr + 4);
      o[0] = (_Float16)(a0.x + b0.x); o[1] = (_Float16)(a0.y + b0.y);
      o[2] = (_Float16)(a0.z + b0.z); o[3] = (_Float16)(a0.w + b0.w);
      o[4] = (_Float16)(a1.x + b1.x); o[5] = (_Float16)(a1.y + b1.y);
      o[6] = (_Float16)(a1.z + b1.z); o[7] = (_Float16)(a1.w + b1.w);
    }
    *(v8h*)&ldsA[(kt * 32 + row + 16 * hiflag) * 16 + eoff] = o;
  }
  __syncthreads();
  int lane = tid & 31, wave = tid >> 5;
  int mh = (lane >> 4) * 8;
  const v16h* pA = (const v16h*)ldsA;
  const v16h* pB = (const v16h*)wn_pack;
  for (int u = 0; u < 2; u++) {
    int nt = wave * 2 + u;
    v8f acc = {};
#pragma unroll
    for (int kt = 0; kt < 8; kt++) {
      v16h a = pA[kt * 32 + lane];
      v16h b = pB[(nt * 8 + kt) * 32 + lane];
      acc = wmma16(a, b, acc);
    }
    int col = nt * 16 + (lane & 15);
#pragma unroll
    for (int j = 0; j < 8; j++) {
      int m = mh + j;
      if (e0 + m < E) atomicAdd(agg + (size_t)s_d[m] * HDIM + col, acc[j]);
    }
  }
}

// ----------------------------------------------------------------- node GEMM
// loop = h@W_evolve + (indeg>0) * h@(W_loop - W_evolve); node = agg/indeg + loop;
// rrelu; cur = l2norm(node). Block: 16 entity rows.
__global__ void node_kernel(const float* h, const float* agg, const float* indeg,
                            const _Float16* __restrict__ we_pack,
                            const _Float16* __restrict__ wd_pack,
                            float* cur, int N) {
  __shared__ __align__(32) _Float16 ldsA[16 * HDIM];
  __shared__ float ndt[16 * HDIM];
  __shared__ float s_inv[16], s_pos[16], red[256], s_invn[16];
  int tid = threadIdx.x;
  int r0 = blockIdx.x * 16;
  if (tid < 16) {
    int rr = r0 + tid;
    float dg = (rr < N) ? indeg[rr] : 0.f;
    s_pos[tid] = dg > 0.f ? 1.f : 0.f;
    s_inv[tid] = dg > 0.f ? 1.f / dg : 0.f;
  }
  for (int i = 0; i < 2; i++) {
    int task = tid * 2 + i;
    int row = task >> 5, g = task & 31;
    int kt = g >> 2, sub = g & 3;
    int kbase = kt * 32 + sub * 8;
    int hiflag = sub & 1, eoff = (sub >> 1) * 8;
    int rr = r0 + row;
    if (rr >= N) rr = N - 1;
    const float* p = h + (size_t)rr * HDIM + kbase;
    float4 a0 = *(const float4*)p, a1 = *(const float4*)(p + 4);
    v8h o;
    o[0] = (_Float16)a0.x; o[1] = (_Float16)a0.y;
    o[2] = (_Float16)a0.z; o[3] = (_Float16)a0.w;
    o[4] = (_Float16)a1.x; o[5] = (_Float16)a1.y;
    o[6] = (_Float16)a1.z; o[7] = (_Float16)a1.w;
    *(v8h*)&ldsA[(kt * 32 + row + 16 * hiflag) * 16 + eoff] = o;
  }
  __syncthreads();
  int lane = tid & 31, wave = tid >> 5;
  int mh = (lane >> 4) * 8;
  const v16h* pA = (const v16h*)ldsA;
  const v16h* pE = (const v16h*)we_pack;
  const v16h* pD = (const v16h*)wd_pack;
  for (int u = 0; u < 2; u++) {
    int nt = wave * 2 + u;
    v8f aE = {}, aD = {};
#pragma unroll
    for (int kt = 0; kt < 8; kt++) {
      v16h a = pA[kt * 32 + lane];
      aE = wmma16(a, pE[(nt * 8 + kt) * 32 + lane], aE);
      aD = wmma16(a, pD[(nt * 8 + kt) * 32 + lane], aD);
    }
    int col = nt * 16 + (lane & 15);
#pragma unroll
    for (int j = 0; j < 8; j++) {
      int m = mh + j;
      int rr = r0 + m;
      if (rr >= N) rr = N - 1;
      float v = agg[(size_t)rr * HDIM + col] * s_inv[m] + aE[j] + s_pos[m] * aD[j];
      v = v >= 0.f ? v : v * RRELU_SLOPE;
      ndt[m * HDIM + col] = v;
    }
  }
  __syncthreads();
  {
    int row = tid >> 4, seg = tid & 15;
    float s = 0.f;
    for (int q = 0; q < 16; q++) {
      float x = ndt[row * HDIM + seg * 16 + q];
      s = fmaf(x, x, s);
    }
    red[tid] = s;
  }
  __syncthreads();
  if (tid < 16) {
    float s = 0.f;
    for (int q = 0; q < 16; q++) s += red[tid * 16 + q];
    s_invn[tid] = 1.f / fmaxf(sqrtf(s), 1e-12f);
  }
  __syncthreads();
  {
    int row = tid >> 4, seg = tid & 15;
    if (r0 + row < N) {
      float iv = s_invn[row];
      for (int q = 0; q < 16; q++)
        cur[(size_t)(r0 + row) * HDIM + seg * 16 + q] =
            ndt[row * HDIM + seg * 16 + q] * iv;
    }
  }
}

// ----------------------------------------------------------------- gate GEMM
// gate = sigmoid(cur@Wt + b); h = gate*cur + (1-gate)*h; hist_t = h.
__global__ void gate_kernel(const float* cur, const _Float16* __restrict__ tg_pack,
                            const float* tgb, float* h, float* hist_t, int N) {
  __shared__ __align__(32) _Float16 ldsA[16 * HDIM];
  int tid = threadIdx.x;
  int r0 = blockIdx.x * 16;
  for (int i = 0; i < 2; i++) {
    int task = tid * 2 + i;
    int row = task >> 5, g = task & 31;
    int kt = g >> 2, sub = g & 3;
    int kbase = kt * 32 + sub * 8;
    int hiflag = sub & 1, eoff = (sub >> 1) * 8;
    int rr = r0 + row;
    if (rr >= N) rr = N - 1;
    const float* p = cur + (size_t)rr * HDIM + kbase;
    float4 a0 = *(const float4*)p, a1 = *(const float4*)(p + 4);
    v8h o;
    o[0] = (_Float16)a0.x; o[1] = (_Float16)a0.y;
    o[2] = (_Float16)a0.z; o[3] = (_Float16)a0.w;
    o[4] = (_Float16)a1.x; o[5] = (_Float16)a1.y;
    o[6] = (_Float16)a1.z; o[7] = (_Float16)a1.w;
    *(v8h*)&ldsA[(kt * 32 + row + 16 * hiflag) * 16 + eoff] = o;
  }
  __syncthreads();
  int lane = tid & 31, wave = tid >> 5;
  int mh = (lane >> 4) * 8;
  const v16h* pA = (const v16h*)ldsA;
  const v16h* pT = (const v16h*)tg_pack;
  for (int u = 0; u < 2; u++) {
    int nt = wave * 2 + u;
    v8f acc = {};
#pragma unroll
    for (int kt = 0; kt < 8; kt++) {
      v16h a = pA[kt * 32 + lane];
      acc = wmma16(a, pT[(nt * 8 + kt) * 32 + lane], acc);
    }
    int col = nt * 16 + (lane & 15);
    float bias = tgb[col];
#pragma unroll
    for (int j = 0; j < 8; j++) {
      int m = mh + j;
      if (r0 + m < N) {
        size_t off = (size_t)(r0 + m) * HDIM + col;
        float gte = sigm(acc[j] + bias);
        float c = cur[off], hv = h[off];
        float hn = fmaf(gte, c - hv, hv);  // g*c + (1-g)*h
        h[off] = hn;
        hist_t[off] = hn;
      }
    }
  }
}

// ------------------------------------------------------------------- launch
extern "C" void kernel_launch(void* const* d_in, const int* in_sizes, int n_in,
                              void* d_out, int out_size, void* d_ws, size_t ws_size,
                              hipStream_t stream) {
  const int* src = (const int*)d_in[0];
  const int* dst = (const int*)d_in[1];
  const int* etype = (const int*)d_in[2];
  const int* rei = (const int*)d_in[3];
  const int* rer = (const int*)d_in[4];
  const float* dyn = (const float*)d_in[5];
  const float* emb_rel = (const float*)d_in[6];
  const float* w_ih = (const float*)d_in[7];
  const float* w_hh = (const float*)d_in[8];
  const float* b_ih = (const float*)d_in[9];
  const float* b_hh = (const float*)d_in[10];
  const float* w_neighbor = (const float*)d_in[11];
  const float* w_loop = (const float*)d_in[12];
  const float* w_evolve = (const float*)d_in[13];
  const float* tg_w = (const float*)d_in[14];
  const float* tg_b = (const float*)d_in[15];
  float* hist = (float*)d_out;

  const int H = in_sizes[15];            // 256
  const int N = in_sizes[5] / H;         // 50000
  const int R2 = in_sizes[6] / H;        // 512
  const long NH = (long)N * H;
  const int T = (int)((long)out_size / NH);  // 8
  const int E = in_sizes[0] / T;         // 200000
  const int M = in_sizes[3] / T;         // 200000

  float* ws = (float*)d_ws;
  float* h = ws;        ws += NH;
  float* cur = ws;      ws += NH;
  float* agg = ws;      ws += NH;
  float* indeg = ws;    ws += N;
  float* h0 = ws;       ws += (long)R2 * H;
  float* xsum = ws;     ws += (long)R2 * H;
  float* cnt = ws;      ws += R2;
  uintptr_t pal = ((uintptr_t)ws + 31) & ~(uintptr_t)31;
  _Float16* wn_pack = (_Float16*)pal;
  _Float16* we_pack = wn_pack + (size_t)H * H;
  _Float16* wd_pack = we_pack + (size_t)H * H;
  _Float16* tg_pack = wd_pack + (size_t)H * H;

  const int B = 256;
  int packBlocks = (H * H + B - 1) / B;
  pack_b_kernel<<<packBlocks, B, 0, stream>>>(w_neighbor, nullptr, wn_pack);
  pack_b_kernel<<<packBlocks, B, 0, stream>>>(w_evolve, nullptr, we_pack);
  pack_b_kernel<<<packBlocks, B, 0, stream>>>(w_loop, w_evolve, wd_pack);  // diff
  pack_b_kernel<<<packBlocks, B, 0, stream>>>(tg_w, nullptr, tg_pack);

  l2norm_rows_kernel<<<N, B, 0, stream>>>(h, dyn);           // h = l2norm(dyn)
  long rh = (long)R2 * H;
  copy_kernel<<<(int)((rh + B - 1) / B), B, 0, stream>>>(h0, emb_rel, rh);

  for (int t = 0; t < T; t++) {
    const int* s_t = src + (long)t * E;
    const int* d_t = dst + (long)t * E;
    const int* e_t = etype + (long)t * E;
    const int* rei_t = rei + (long)t * M;
    const int* rer_t = rer + (long)t * M;

    zero_kernel<<<(int)((rh + B - 1) / B), B, 0, stream>>>(xsum, rh);
    zero_kernel<<<(R2 + B - 1) / B, B, 0, stream>>>(cnt, R2);
    zero_kernel<<<(int)((NH + B - 1) / B), B, 0, stream>>>(agg, NH);
    zero_kernel<<<(N + B - 1) / B, B, 0, stream>>>(indeg, N);

    long rsT = (long)M * 64;
    rel_scatter_kernel<<<(int)((rsT + B - 1) / B), B, 0, stream>>>(
        rei_t, rer_t, h, xsum, cnt, M);
    gru_kernel<<<R2, B, 0, stream>>>(emb_rel, xsum, cnt, w_ih, w_hh, b_ih, b_hh, h0);
    indeg_kernel<<<(E + B - 1) / B, B, 0, stream>>>(d_t, indeg, E);
    msg_kernel<<<(E + 15) / 16, B, 0, stream>>>(s_t, d_t, e_t, h, h0, wn_pack, agg, E);
    node_kernel<<<(N + 15) / 16, B, 0, stream>>>(h, agg, indeg, we_pack, wd_pack, cur, N);
    gate_kernel<<<(N + 15) / 16, B, 0, stream>>>(cur, tg_pack, tg_b, h,
                                                 hist + (long)t * NH, N);
  }
}
